// LSTMTZ0InitializerV2_49761491092070
// MI455X (gfx1250) — matrix operand order repaired and checked by reading, not compiled, not run on hardware
//
#include <hip/hip_runtime.h>
#include <hip/hip_bf16.h>

// Problem constants (fixed by the reference)
#define TB 4096   // batch
#define TT 365    // time steps
#define TF 16     // input features
#define THID 256  // hidden
#define KP 288    // padded K = 256 (h) + 16 (x) + 16 (zero pad)
#define KC 9      // K chunks of 32
#define NTILE 8   // 16-wide N tiles per wave (8 tiles = 128 gate cols)
#define BT 16     // batch rows per workgroup

typedef __attribute__((ext_vector_type(16))) __bf16 v16bf;
typedef __attribute__((ext_vector_type(8)))  __bf16 v8bf;
typedef __attribute__((ext_vector_type(8)))  float  v8f;

__device__ __forceinline__ float sigmoidf_(float x) {
  return 1.0f / (1.0f + __expf(-x));
}

// ---------------------------------------------------------------------------
// Prologue: pack [W_hh | W_ih | 0] (bf16, K=288) into WMMA-B-fragment-major
// tiles. Physical tile p = wn*8 + gate*2 + jl covers logical gate columns
// col = gate*256 + (wn*2 + jl)*16 + n  (gate-interleaved so each wave owns
// matching i/f/g/o blocks for a contiguous 32-wide hidden slice).
// Fragment layout per (p,kc) tile: lane<16 -> K=kb..kb+15, N=lane;
// lane>=16 -> K=kb+16..kb+31, N=lane-16; 16 bf16 (32B) contiguous per lane.
// ---------------------------------------------------------------------------
__global__ void pack_weights_kernel(const float* __restrict__ W_ih,
                                    const float* __restrict__ W_hh,
                                    __bf16* __restrict__ packedB) {
  int pk   = blockIdx.x;            // 0..575  (= p*9 + kc)
  int p    = pk / KC;
  int kc   = pk % KC;
  int lane = threadIdx.x;           // 0..31
  int wn   = p >> 3;
  int t_   = p & 7;
  int gate = t_ >> 1;
  int jl   = t_ & 1;
  int n    = lane & 15;
  int col  = gate * 256 + (wn * 2 + jl) * 16 + n;
  int kbase = kc * 32 + ((lane >> 4) << 4);   // +0 or +16
  __bf16* dst = packedB + (size_t)pk * 512 + lane * 16;
#pragma unroll
  for (int e = 0; e < 16; ++e) {
    int k = kbase + e;
    float v;
    if (k < 256)      v = W_hh[col * 256 + k];
    else if (k < 272) v = W_ih[col * 16 + (k - 256)];
    else              v = 0.0f;
    dst[e] = (__bf16)v;
  }
}

// ---------------------------------------------------------------------------
// Main LSTM kernel: one WG = 16 batch rows, 8 waves. Wave wn owns gate
// columns for hidden units j in [wn*32, wn*32+32) (all four gates).
// A = [h_t | x_t | 0] (16 x 288 bf16) lives in LDS; c stays in registers.
// Per step: 72 v_wmma_f32_16x16x32_bf16 per wave; weights stream from L2
// (576 KB packed buffer is permanently L2-resident; 192 MB L2).
// ---------------------------------------------------------------------------
__global__ void __launch_bounds__(256) lstm_kernel(
    const float*  __restrict__ w,        // [4096, 365, 16]
    const __bf16* __restrict__ packedB,  // 64 tiles x 9 chunks x 1KB
    const float*  __restrict__ b_ih,
    const float*  __restrict__ b_hh,
    float*        __restrict__ out_h) {  // [4096, 256]
  __shared__ __align__(16) __bf16 A[BT * KP];

  const int tid   = threadIdx.x;
  const int lane  = tid & 31;
  const int wn    = tid >> 5;            // wave id 0..7
  const int bbase = blockIdx.x * BT;

  // zero A (h0 = 0 and the K-pad region)
  for (int i = tid; i < (BT * KP) / 2; i += 256) ((unsigned*)A)[i] = 0u;

  // preload per-column bias (permuted column order matches packing)
  float biasv[NTILE];
#pragma unroll
  for (int t_ = 0; t_ < NTILE; ++t_) {
    int gate = t_ >> 1, jl = t_ & 1;
    int col  = gate * 256 + (wn * 2 + jl) * 16 + (lane & 15);
    biasv[t_] = b_ih[col] + b_hh[col];
  }

  // stage x_0 into A columns 256..271
  {
    int m = tid >> 4, f = tid & 15;
    float x = w[((size_t)(bbase + m) * TT + 0) * TF + f];
    __syncthreads();                     // zeroing done before x store
    A[m * KP + 256 + f] = (__bf16)x;
  }
  __syncthreads();

  float creg[2][8];
  float hv[2][8];
#pragma unroll
  for (int jl = 0; jl < 2; ++jl)
#pragma unroll
    for (int r = 0; r < 8; ++r) { creg[jl][r] = 0.0f; hv[jl][r] = 0.0f; }

  const int mrow     = lane & 15;               // A-matrix M = lane%16
  const int koff_sel = (lane & 16) ? 16 : 0;    // A-layout K half select

  // wave-uniform weight-slice base, forced scalar for saddr-form loads
  const unsigned wave_byte_off =
      (unsigned)__builtin_amdgcn_readfirstlane((int)(wn * (NTILE * KC * 512 * 2)));
  const char* bwave_c = (const char*)packedB + wave_byte_off;

  for (int t = 0; t < TT; ++t) {
    // Defeat LICM: a uniform scalar redefined by volatile asm each iteration
    // makes the weight-load addresses loop-variant, so the compiler cannot
    // hoist the 144 fragment loads out of the time loop (which previously
    // caused massive scratch spills). We *want* a fresh L2 stream per step.
    unsigned lic = 0;
    asm volatile("" : "+s"(lic));
    const __bf16* bptr = (const __bf16*)(bwave_c + lic);

    // init accumulators with bias (broadcast over M rows)
    v8f acc[NTILE];
#pragma unroll
    for (int nt = 0; nt < NTILE; ++nt) {
      v8f a;
#pragma unroll
      for (int r = 0; r < 8; ++r) a[r] = biasv[nt];
      acc[nt] = a;
    }

    // ---- software-pipelined WMMA: load B(kc+1) while computing on B(kc) ----
    v16bf bcur[NTILE];
#pragma unroll
    for (int nt = 0; nt < NTILE; ++nt)
      bcur[nt] = *(const v16bf*)(bptr + (size_t)(nt * KC) * 512 + lane * 16);

#pragma unroll
    for (int kc = 0; kc < KC; ++kc) {
      v16bf bnxt[NTILE];
      if (kc + 1 < KC) {
#pragma unroll
        for (int nt = 0; nt < NTILE; ++nt)
          bnxt[nt] = *(const v16bf*)(bptr + (size_t)(nt * KC + kc + 1) * 512 + lane * 16);
      }

      int koff = kc * 32 + koff_sel;
      v8bf alo = *(const v8bf*)&A[mrow * KP + koff];        // ds_load_b128
      v8bf ahi = *(const v8bf*)&A[mrow * KP + koff + 8];    // ds_load_b128
      v16bf af = __builtin_shufflevector(alo, ahi,
                   0,1,2,3,4,5,6,7,8,9,10,11,12,13,14,15);

#pragma unroll
      for (int nt = 0; nt < NTILE; ++nt)
        acc[nt] = __builtin_amdgcn_wmma_f32_16x16x32_bf16(
            false, af, false, bcur[nt], (short)0, acc[nt], false, false);

#pragma unroll
      for (int nt = 0; nt < NTILE; ++nt) bcur[nt] = bnxt[nt];  // SSA-renamed away
    }

    // nonlinearity + state update, fully in registers (i,f,g,o co-located)
#pragma unroll
    for (int jl = 0; jl < 2; ++jl) {
#pragma unroll
      for (int r = 0; r < 8; ++r) {
        float ig = sigmoidf_(acc[0 * 2 + jl][r]);
        float fg = sigmoidf_(acc[1 * 2 + jl][r]);
        float gg = tanhf   (acc[2 * 2 + jl][r]);
        float og = sigmoidf_(acc[3 * 2 + jl][r]);
        float cn = fg * creg[jl][r] + ig * gg;
        creg[jl][r] = cn;
        hv[jl][r]   = og * tanhf(cn);
      }
    }

    // prefetch x_{t+1} from HBM while gates settle
    float xnext = 0.0f;
    if (t + 1 < TT) {
      int m = tid >> 4, f = tid & 15;
      xnext = w[((size_t)(bbase + m) * TT + (t + 1)) * TF + f];
    }

    __syncthreads();  // all waves finished reading A (h_{t-1}, x_t)

    // write h_t (bf16) into A columns 0..255 per C/D-matrix layout
#pragma unroll
    for (int jl = 0; jl < 2; ++jl) {
      int j = wn * 32 + jl * 16 + (lane & 15);
#pragma unroll
      for (int r = 0; r < 8; ++r) {
        int m = r + ((lane >> 4) << 3);
        A[m * KP + j] = (__bf16)hv[jl][r];
      }
    }
    if (t + 1 < TT) {
      int m = tid >> 4, f = tid & 15;
      A[m * KP + 256 + f] = (__bf16)xnext;
    }
    __syncthreads();
  }

  // final h in f32 straight from registers
#pragma unroll
  for (int jl = 0; jl < 2; ++jl) {
    int j = wn * 32 + jl * 16 + (lane & 15);
#pragma unroll
    for (int r = 0; r < 8; ++r) {
      int m = r + ((lane >> 4) << 3);
      out_h[(size_t)(bbase + m) * THID + j] = hv[jl][r];
    }
  }
}

// ---------------------------------------------------------------------------
// Head: tz = ELU(h) @ W_out.T + b_out  (tiny: 4096 x 256 x 2)
// ---------------------------------------------------------------------------
__global__ void head_kernel(const float* __restrict__ h,
                            const float* __restrict__ W_out,
                            const float* __restrict__ b_out,
                            float* __restrict__ out01) {
  int b = blockIdx.x * blockDim.x + threadIdx.x;
  if (b >= TB) return;
  const float* hr = h + (size_t)b * THID;
  float s0 = 0.0f, s1 = 0.0f;
#pragma unroll 4
  for (int j = 0; j < THID; ++j) {
    float x = hr[j];
    float e = (x > 0.0f) ? x : (__expf(x) - 1.0f);
    s0 += e * W_out[j];
    s1 += e * W_out[THID + j];
  }
  out01[b]      = s0 + b_out[0];
  out01[TB + b] = s1 + b_out[1];
}

extern "C" void kernel_launch(void* const* d_in, const int* in_sizes, int n_in,
                              void* d_out, int out_size, void* d_ws, size_t ws_size,
                              hipStream_t stream) {
  const float* w     = (const float*)d_in[0];
  const float* W_ih  = (const float*)d_in[1];
  const float* W_hh  = (const float*)d_in[2];
  const float* b_ih  = (const float*)d_in[3];
  const float* b_hh  = (const float*)d_in[4];
  const float* W_out = (const float*)d_in[5];
  const float* b_out = (const float*)d_in[6];
  float*  out     = (float*)d_out;            // [tz0_col0 | tz0_col1 | h]
  __bf16* packedB = (__bf16*)d_ws;            // 64*9*1KB = 576 KB

  // 1) pack weights (bf16, fragment-major, gate-interleaved columns)
  pack_weights_kernel<<<64 * KC, 32, 0, stream>>>(W_ih, W_hh, packedB);

  // 2) LSTM recurrence: 256 WGs x 8 waves, WMMA bf16, L2-resident weights
  lstm_kernel<<<TB / BT, 256, 0, stream>>>(w, packedB, b_ih, b_hh, out + 2 * TB);

  // 3) ELU -> Linear(2) head
  head_kernel<<<(TB + 255) / 256, 256, 0, stream>>>(out + 2 * TB, W_out, b_out, out);
}
